// MultiAgents_3770981286789
// MI455X (gfx1250) — matrix operand order repaired and checked by reading, not compiled, not run on hardware
//
#include <hip/hip_runtime.h>

// ---------------------------------------------------------------------------
// Fused CapsNet multi-agent kernel for gfx1250 (CDNA5, wave32, WMMA).
// - All three GEMMs: v_wmma_f32_16x16x32_bf16 with bf16 hi/lo error
//   compensation (3 WMMAs / K=32 chunk) -> fp32-class accuracy at bf16 rate.
// - Weights are pre-split ONCE into packed bf16-pair arrays in d_ws by a
//   prepack kernel (~1.2 MB); main-kernel weight staging is a pure b128 copy.
// - Weight tiles stream into LDS via global_load_async_to_lds_b128 (ASYNCcnt),
//   double-buffered so copies overlap WMMA compute.
// - Block = 256 threads (8 waves) owns 32 batch rows; h, u, u_hat all in LDS
//   (307 KB), so HBM traffic is just x + out (+L2-resident weights).
// ---------------------------------------------------------------------------

typedef __bf16 v16bf __attribute__((ext_vector_type(16)));
typedef float  v8f   __attribute__((ext_vector_type(8)));

union FragU {
  v16bf v;
  unsigned int u[8];
};

__device__ __forceinline__ unsigned short bf16_rne(float f) {
  unsigned int u = __float_as_uint(f);
  u += 0x7FFFu + ((u >> 16) & 1u);
  return (unsigned short)(u >> 16);
}

// Split two floats into packed bf16-hi pair and bf16-lo (residual) pair.
__device__ __forceinline__ void split_pack(float f0, float f1,
                                           unsigned int& ph, unsigned int& pl) {
  unsigned short h0 = bf16_rne(f0);
  unsigned short h1 = bf16_rne(f1);
  float r0 = f0 - __uint_as_float((unsigned int)h0 << 16);
  float r1 = f1 - __uint_as_float((unsigned int)h1 << 16);
  unsigned short l0 = bf16_rne(r0);
  unsigned short l1 = bf16_rne(r1);
  ph = (unsigned int)h0 | ((unsigned int)h1 << 16);
  pl = (unsigned int)l0 | ((unsigned int)l1 << 16);
}

__device__ __forceinline__ v8f zero_v8f() {
  v8f z;
#pragma unroll
  for (int i = 0; i < 8; ++i) z[i] = 0.0f;
  return z;
}

__device__ __forceinline__ v8f wmma_bf16(const FragU& A, const FragU& B, v8f C) {
  return __builtin_amdgcn_wmma_f32_16x16x32_bf16(false, A.v, false, B.v,
                                                 (short)0, C, false, false);
}

__device__ __forceinline__ float wsum32(float x) {
#pragma unroll
  for (int o = 16; o > 0; o >>= 1) x += __shfl_xor(x, o, 32);
  return x;
}

// ---- async copy global->LDS (CDNA5 ASYNCcnt path), guarded ----------------
#ifdef __has_builtin
#  if __has_builtin(__builtin_amdgcn_global_load_async_to_lds_b128)
#    define HAVE_ASYNC_LDS 1
#  endif
#  if __has_builtin(__builtin_amdgcn_s_wait_asynccnt)
#    define HAVE_WAIT_ASYNC 1
#  endif
#endif
#ifndef HAVE_ASYNC_LDS
#  define HAVE_ASYNC_LDS 0
#endif
#ifndef HAVE_WAIT_ASYNC
#  define HAVE_WAIT_ASYNC 0
#endif

// The builtin takes pointers to int4 vectors: v4i addrspace(1)* (global src)
// and v4i addrspace(3)* (LDS dst).  (Clang prints AS1 as pointee-level
// "__device__" in HIP mode, per the round-2 diagnostic.)
typedef int v4i_gcc __attribute__((vector_size(16)));
typedef __attribute__((address_space(1))) v4i_gcc as1_v4i;
typedef __attribute__((address_space(3))) v4i_gcc as3_v4i;

__device__ __forceinline__ void cp_lds_b128(unsigned int* lds_dst,
                                            const unsigned int* gsrc) {
#if HAVE_ASYNC_LDS
  as1_v4i* g = reinterpret_cast<as1_v4i*>((unsigned long long)gsrc);
  as3_v4i* l = reinterpret_cast<as3_v4i*>((unsigned int)(unsigned long long)lds_dst);
  __builtin_amdgcn_global_load_async_to_lds_b128(g, l, 0, 0);
#else
  *reinterpret_cast<uint4*>(lds_dst) = *reinterpret_cast<const uint4*>(gsrc);
#endif
}

__device__ __forceinline__ void wait_async_copies() {
#if HAVE_ASYNC_LDS
#  if HAVE_WAIT_ASYNC
  __builtin_amdgcn_s_wait_asynccnt(0);
#  else
  asm volatile("s_wait_asynccnt 0" ::: "memory");
#  endif
#endif
}

// ---- workspace layout (u32 units): pre-split packed bf16-pair weights -----
constexpr int WS_W1H = 0;                    // [400 kk][256 n] (kk 392..399 = 0)
constexpr int WS_W1L = WS_W1H + 400 * 256;   // 102400
constexpr int WS_W2H = WS_W1L + 400 * 256;   // [32 kk][256 n]
constexpr int WS_W2L = WS_W2H + 32 * 256;
constexpr int WS_WRH = WS_W2L + 32 * 256;    // [8 (a,hf)][32 kk][160 c]
constexpr int WS_WRL = WS_WRH + 8 * 32 * 160;
constexpr int WS_TOTAL_U32 = WS_WRL + 8 * 32 * 160;  // 303104 u32 = 1.16 MB

// ---- LDS layout (u32 units) ------------------------------------------------
constexpr int HPH  = 0;                   // h packed hi: [32 m][129]
constexpr int HPL  = HPH + 32 * 129;
constexpr int UPH  = HPL + 32 * 129;      // u packed hi
constexpr int UPL  = UPH + 32 * 129;
constexpr int UHAT = UPL + 32 * 129;      // f32 [32][1284] (also u_pre scratch)
constexpr int DREG = UHAT + 32 * 1284;    // 57600: reused staging region
// GEMM1 ping-pong: AT[p] = {hi[32*17], lo[32*17]}, BT[p] = {hi[16*260], lo[16*260]}
constexpr int AT0  = DREG;                // + p*1088 ; lo at +544
constexpr int BT0  = DREG + 2 * 1088;     // + p*8320 ; lo at +4160
// GEMM2: B2 = {hi[32*260], lo[32*260]}
constexpr int B2H  = DREG;
constexpr int B2L  = DREG + 32 * 260;
// GEMM3 ping-pong: B3[p] = {hi[32*164], lo[32*164]}
constexpr int B30  = DREG;                // + p*10496 ; lo at +5248
constexpr int SMEM_U32 = DREG + 2 * 10496;  // 78592 u32 = 314368 B (307 KB)

// ===========================================================================
// Prepack kernel: split W1 / W2 / W into packed bf16 hi/lo pair arrays.
// ===========================================================================
__global__ __launch_bounds__(256) void caps_prepack_kernel(
    const float* __restrict__ W1, const float* __restrict__ W2,
    const float* __restrict__ WR, unsigned int* __restrict__ ws)
{
  int idx = blockIdx.x * 256 + threadIdx.x;
  if (idx < 102400) {                                  // W1: kk-major [400][256]
    int kkg = idx >> 8, n = idx & 255;
    float f0 = 0.0f, f1 = 0.0f;
    if (kkg < 392) {
      int a = n >> 6, o = n & 63, k = 2 * kkg;
      const float* p = W1 + (size_t)a * 50176 + (size_t)k * 64 + o;
      f0 = p[0];
      f1 = p[64];
    }
    unsigned int ph, pl;
    split_pack(f0, f1, ph, pl);
    ws[WS_W1H + idx] = ph;
    ws[WS_W1L + idx] = pl;
  } else if (idx < 110592) {                           // W2: [32][256]
    int e = idx - 102400;
    int kk = e >> 8, n = e & 255;
    int a = n >> 6, o = n & 63;
    const float* p = W2 + a * 4096 + 2 * kk * 64 + o;
    unsigned int ph, pl;
    split_pack(p[0], p[64], ph, pl);
    ws[WS_W2H + e] = ph;
    ws[WS_W2L + e] = pl;
  } else if (idx < 151552) {                           // W: [8][32][160]
    int e = idx - 110592;
    int r = e / 5120, w = e % 5120;
    int kk = w / 160, c = w % 160;
    int C = (r >> 1) * 320 + (r & 1) * 160 + c;        // a*320 + hf*160 + c
    const float2 p = *reinterpret_cast<const float2*>(WR + (size_t)C * 64 + 2 * kk);
    unsigned int ph, pl;
    split_pack(p.x, p.y, ph, pl);
    ws[WS_WRH + e] = ph;
    ws[WS_WRL + e] = pl;
  }
}

// ===========================================================================
// Main fused kernel.
// ===========================================================================
__global__ __launch_bounds__(256, 1) void caps_fused_kernel(
    const float* __restrict__ X,  const float* __restrict__ B1v,
    const float* __restrict__ B2v, const unsigned int* __restrict__ ws,
    float* __restrict__ OUT)
{
  __shared__ __align__(16) unsigned int sm[SMEM_U32];
  float* smf = reinterpret_cast<float*>(sm);

  const int tid  = threadIdx.x;
  const int wave = tid >> 5;
  const int lane = tid & 31;
  const int g    = lane >> 4;     // lane-half: K group (A/B), M+8 (C/D)
  const int sl   = lane & 15;
  const int row0 = blockIdx.x * 32;
  const int mt   = wave & 1;      // m-tile 0..1
  const int n0   = wave >> 1;     // base n-tile 0..3
  const int mrow = mt * 16 + sl;  // A-fragment row

  const unsigned int* w1h = ws + WS_W1H;
  const unsigned int* w1l = ws + WS_W1L;

  // ===================== GEMM1: h = relu(x @ W1 + b1) ======================
  auto stage1 = [&](int ch, int p) {
    const int atH = AT0 + p * 1088, atL = atH + 544;
    const int kb = ch * 32;
    // x tile: convert+split in-kernel (cheap: 2 entries/thread)
#pragma unroll
    for (int s = 0; s < 2; ++s) {
      int e = tid + 256 * s;
      int m = e >> 4, kk = e & 15;
      int k = kb + 2 * kk;
      float f0 = 0.0f, f1 = 0.0f;
      if (k < 784) {                       // 784 even -> pair fully valid
        const float2 pr = *reinterpret_cast<const float2*>(
            X + (size_t)(row0 + m) * 784 + k);
        f0 = pr.x; f1 = pr.y;
      }
      unsigned int ph, pl;
      split_pack(f0, f1, ph, pl);
      sm[atH + m * 17 + kk] = ph;
      sm[atL + m * 17 + kk] = pl;
    }
    // W1 tile: pure b128 copy of pre-split pairs (async when available)
    const int btH = BT0 + p * 8320, btL = btH + 4160;
    const unsigned int* gh = w1h + ch * 4096;
    const unsigned int* gl = w1l + ch * 4096;
#pragma unroll
    for (int s = 0; s < 4; ++s) {
      int e = tid + 256 * s;
      int kk = e >> 6, c4 = (e & 63) << 2;
      cp_lds_b128(&sm[btH + kk * 260 + c4], gh + kk * 256 + c4);
      cp_lds_b128(&sm[btL + kk * 260 + c4], gl + kk * 256 + c4);
    }
  };

  v8f acc[4];
#pragma unroll
  for (int t = 0; t < 4; ++t) acc[t] = zero_v8f();

  stage1(0, 0);
  wait_async_copies();
  __syncthreads();

  for (int ch = 0; ch < 25; ++ch) {        // K = 784 = 24*32 + 16 (padded)
    const int p = ch & 1;
    if (ch + 1 < 25) stage1(ch + 1, p ^ 1);    // prefetch next tiles

    const int atH = AT0 + p * 1088, atL = atH + 544;
    const int btH = BT0 + p * 8320, btL = btH + 4160;

    FragU AH, AL;
#pragma unroll
    for (int v = 0; v < 8; ++v) {
      int kk = 8 * g + v;                          // contiguous k = 16g + s (matches B)
      AH.u[v] = sm[atH + mrow * 17 + kk];
      AL.u[v] = sm[atL + mrow * 17 + kk];
    }
#pragma unroll
    for (int t = 0; t < 4; ++t) {
      int n = (n0 + 4 * t) * 16 + sl;
      FragU BH, BL;
#pragma unroll
      for (int v = 0; v < 8; ++v) {                // contiguous k = 16g + s (matches A)
        BH.u[v] = sm[btH + (8 * g + v) * 260 + n];
        BL.u[v] = sm[btL + (8 * g + v) * 260 + n];
      }
      acc[t] = wmma_bf16(AH, BH, acc[t]);          // hi*hi
      acc[t] = wmma_bf16(AH, BL, acc[t]);          // hi*lo
      acc[t] = wmma_bf16(AL, BH, acc[t]);          // lo*hi
    }
    wait_async_copies();                           // next tiles landed
    __syncthreads();
  }

  // bias + relu, repack straight to bf16 hi/lo h (pair adjacent n via shfl)
#pragma unroll
  for (int t = 0; t < 4; ++t) {
    int n = (n0 + 4 * t) * 16 + sl;
    float bv = B1v[n];
#pragma unroll
    for (int r = 0; r < 8; ++r) {
      float val = fmaxf(acc[t][r] + bv, 0.0f);
      float oth = __shfl_xor(val, 1, 32);
      if ((sl & 1) == 0) {
        int m = mt * 16 + r + 8 * g;               // C/D row mapping
        unsigned int ph, pl;
        split_pack(val, oth, ph, pl);
        sm[HPH + m * 129 + (n >> 1)] = ph;
        sm[HPL + m * 129 + (n >> 1)] = pl;
      }
    }
  }

  // ============== GEMM2: u = squash(h @ W2 + b2)  (block-diag) =============
  {
    const unsigned int* gh = ws + WS_W2H;
    const unsigned int* gl = ws + WS_W2L;
#pragma unroll
    for (int s = 0; s < 8; ++s) {
      int e = tid + 256 * s;
      int kk = e >> 6, c4 = (e & 63) << 2;
      cp_lds_b128(&sm[B2H + kk * 260 + c4], gh + kk * 256 + c4);
      cp_lds_b128(&sm[B2L + kk * 260 + c4], gl + kk * 256 + c4);
    }
  }
  wait_async_copies();
  __syncthreads();

  v8f acc2[4];
#pragma unroll
  for (int t = 0; t < 4; ++t) acc2[t] = zero_v8f();
#pragma unroll
  for (int t = 0; t < 4; ++t) {
    int agent = t;                                 // (n0+4t)*16/64 == t
    int n = (n0 + 4 * t) * 16 + sl;
#pragma unroll
    for (int c = 0; c < 2; ++c) {                  // K = 64 -> 2 chunks
      FragU AH, AL, BH, BL;
      int kbA = agent * 32 + c * 16;
#pragma unroll
      for (int v = 0; v < 8; ++v) {
        int kk = kbA + 8 * g + v;
        AH.u[v] = sm[HPH + mrow * 129 + kk];
        AL.u[v] = sm[HPL + mrow * 129 + kk];
        BH.u[v] = sm[B2H + (c * 16 + 8 * g + v) * 260 + n];
        BL.u[v] = sm[B2L + (c * 16 + 8 * g + v) * 260 + n];
      }
      acc2[t] = wmma_bf16(AH, BH, acc2[t]);
      acc2[t] = wmma_bf16(AH, BL, acc2[t]);
      acc2[t] = wmma_bf16(AL, BH, acc2[t]);
    }
  }
  // u_pre (f32) -> UHAT scratch, stride 257 (conflict-free column writes)
#pragma unroll
  for (int t = 0; t < 4; ++t) {
    int n = (n0 + 4 * t) * 16 + sl;
    float bv = B2v[n];
#pragma unroll
    for (int r = 0; r < 8; ++r) {
      int m = mt * 16 + r + 8 * g;
      smf[UHAT + m * 257 + n] = acc2[t][r] + bv;
    }
  }
  __syncthreads();

  // squash over 64 dims per (row, agent); repack u as bf16 hi/lo
  if (tid < 128) {
    int m = tid >> 2, a = tid & 3;
    const float* up = smf + UHAT + m * 257 + a * 64;
    float sq = 0.0f;
#pragma unroll
    for (int k = 0; k < 64; ++k) { float f = up[k]; sq += f * f; }
    float scale = sq / ((1.0f + sq) * sqrtf(sq + 1e-8f));
#pragma unroll
    for (int kk = 0; kk < 32; ++kk) {
      unsigned int ph, pl;
      split_pack(up[2 * kk] * scale, up[2 * kk + 1] * scale, ph, pl);
      sm[UPH + m * 129 + a * 32 + kk] = ph;
      sm[UPL + m * 129 + a * 32 + kk] = pl;
    }
  }
  __syncthreads();

  // ============ GEMM3: u_hat[b,a,j,i] = sum_k W[a,j,i,k] u[b,a,k] ==========
  const unsigned int* wrh = ws + WS_WRH;
  const unsigned int* wrl = ws + WS_WRL;
  auto stage3 = [&](int idx, int p) {
    const int b3H = B30 + p * 10496, b3L = b3H + 5248;
    const unsigned int* gh = wrh + idx * 5120;
    const unsigned int* gl = wrl + idx * 5120;
#pragma unroll
    for (int s = 0; s < 5; ++s) {
      int e = tid + 256 * s;
      int kk = e / 40, c4 = (e % 40) << 2;
      cp_lds_b128(&sm[b3H + kk * 164 + c4], gh + kk * 160 + c4);
      cp_lds_b128(&sm[b3L + kk * 164 + c4], gl + kk * 160 + c4);
    }
  };

  stage3(0, 0);
  wait_async_copies();
  __syncthreads();

  for (int it8 = 0; it8 < 8; ++it8) {              // (agent, half) pairs
    const int a = it8 >> 1, hf = it8 & 1, p = it8 & 1;
    if (it8 + 1 < 8) stage3(it8 + 1, p ^ 1);       // prefetch next W tile

    const int b3H = B30 + p * 10496, b3L = b3H + 5248;
#pragma unroll
    for (int s = 0; s < 3; ++s) {
      int nt = n0 + 4 * s;
      if (nt < 10) {                               // wave-uniform branch
        int n = nt * 16 + sl;
        v8f acc3 = zero_v8f();
#pragma unroll
        for (int c = 0; c < 2; ++c) {
          FragU AH, AL, BH, BL;
          int kbA = a * 32 + c * 16;
#pragma unroll
          for (int v = 0; v < 8; ++v) {
            int kk = kbA + 8 * g + v;
            AH.u[v] = sm[UPH + mrow * 129 + kk];
            AL.u[v] = sm[UPL + mrow * 129 + kk];
            BH.u[v] = sm[b3H + (c * 16 + 8 * g + v) * 164 + n];
            BL.u[v] = sm[b3L + (c * 16 + 8 * g + v) * 164 + n];
          }
          acc3 = wmma_bf16(AH, BH, acc3);
          acc3 = wmma_bf16(AH, BL, acc3);
          acc3 = wmma_bf16(AL, BH, acc3);
        }
#pragma unroll
        for (int r = 0; r < 8; ++r) {
          int m = mt * 16 + r + 8 * g;
          smf[UHAT + m * 1284 + a * 320 + hf * 160 + nt * 16 + sl] = acc3[r];
        }
      }
    }
    wait_async_copies();
    __syncthreads();
  }

  // ================= dynamic routing (3 iters), 1 row per wave =============
#pragma unroll 1
  for (int t = 0; t < 4; ++t) {
    int m = wave + 8 * t;
    const float* ub = smf + UHAT + m * 1284;
    float uh[40];                                  // lane = i (0..31)
#pragma unroll
    for (int q = 0; q < 40; ++q)
      uh[q] = ub[(q / 10) * 320 + (q % 10) * 32 + lane];

    float logit[40];
#pragma unroll
    for (int q = 0; q < 40; ++q) logit[q] = 0.0f;
    float vcap[10];

#pragma unroll 1
    for (int it = 0; it < 3; ++it) {
      float cc[40];
#pragma unroll
      for (int a = 0; a < 4; ++a) {                // softmax over 10 capsules
        float mx = logit[a * 10];
#pragma unroll
        for (int j = 1; j < 10; ++j) mx = fmaxf(mx, logit[a * 10 + j]);
        float se = 0.0f;
#pragma unroll
        for (int j = 0; j < 10; ++j) {
          float e = __expf(logit[a * 10 + j] - mx);
          cc[a * 10 + j] = e;
          se += e;
        }
        float inv = 1.0f / se;
#pragma unroll
        for (int j = 0; j < 10; ++j) cc[a * 10 + j] *= inv;
      }
#pragma unroll
      for (int j = 0; j < 10; ++j) {
        float s = 0.0f;
#pragma unroll
        for (int a = 0; a < 4; ++a) s += cc[a * 10 + j] * uh[a * 10 + j];
        float sq = wsum32(s * s);                  // squash over 32 lanes (= i)
        float scale = sq / ((1.0f + sq) * sqrtf(sq + 1e-8f));
        vcap[j] = s * scale;
      }
      if (it < 2) {
#pragma unroll
        for (int q = 0; q < 40; ++q) {
          float agr = wsum32(uh[q] * vcap[q % 10]);  // agreement = <u_hat, v>
          logit[q] += agr;
        }
      }
    }
    float* ob = OUT + (size_t)(row0 + m) * 320;
#pragma unroll
    for (int j = 0; j < 10; ++j) ob[j * 32 + lane] = vcap[j];
  }
}

extern "C" void kernel_launch(void* const* d_in, const int* in_sizes, int n_in,
                              void* d_out, int out_size, void* d_ws, size_t ws_size,
                              hipStream_t stream) {
  (void)in_sizes; (void)n_in; (void)out_size; (void)ws_size;  // needs ~1.2 MB ws
  const float* x  = (const float*)d_in[0];
  const float* W1 = (const float*)d_in[1];
  const float* b1 = (const float*)d_in[2];
  const float* W2 = (const float*)d_in[3];
  const float* b2 = (const float*)d_in[4];
  const float* Wr = (const float*)d_in[5];
  float* out = (float*)d_out;
  unsigned int* ws = (unsigned int*)d_ws;

  // 1) pre-split weights into packed bf16 hi/lo pairs (151552 entries)
  hipLaunchKernelGGL(caps_prepack_kernel, dim3(592), dim3(256), 0, stream,
                     W1, W2, Wr, ws);
  // 2) fused forward pass: 1024 blocks x 32 rows
  hipLaunchKernelGGL(caps_fused_kernel, dim3(1024), dim3(256), 0, stream,
                     x, b1, b2, ws, out);
}
